// GraphModel_2241972928707
// MI455X (gfx1250) — matrix operand order repaired and checked by reading, not compiled, not run on hardware
//
#include <hip/hip_runtime.h>
#include <hip/hip_bf16.h>

typedef __attribute__((ext_vector_type(16))) _Float16 v16h;
typedef __attribute__((ext_vector_type(8)))  float    v8f;

#define DIMF 128
#define LNEPS 1e-5f

// ---------------------------------------------------------------- utilities
__global__ void k_fill(float* __restrict__ p, float v, int n) {
    int i = blockIdx.x * blockDim.x + threadIdx.x;
    if (i < n) p[i] = v;
}

__global__ void k_degree(const int* __restrict__ dst, float* __restrict__ deg, int E) {
    int e = blockIdx.x * blockDim.x + threadIdx.x;
    if (e < E) atomicAdd(&deg[dst[e]], 1.0f);
}

__global__ void k_rsqrt_inplace(float* __restrict__ d, int n) {
    int i = blockIdx.x * blockDim.x + threadIdx.x;
    if (i < n) d[i] = rsqrtf(d[i]);   // deg >= 1 always (self loop), matches reference
}

// ------------------------------------------------- pack weights into WMMA B-fragment layout (f16)
// Fragment fi = kt*ntilesN + nt. Per lane: 16 f16 (32B contiguous) holding the
// B-matrix 32x16 layout: lane n = nt*16 + (lane&15), K rows krow0..krow0+7, krow0+16..krow0+23.
__global__ void k_pack_b(const float* __restrict__ W, _Float16* __restrict__ Wp,
                         int C, int ntilesN) {
    const int lane = threadIdx.x & 31;
    const int fi   = blockIdx.x * (blockDim.x >> 5) + (threadIdx.x >> 5);
    const int nfrag = 4 * ntilesN;                  // K = 128 -> 4 chunks of 32
    if (fi >= nfrag) return;
    const int kt = fi / ntilesN;
    const int nt = fi % ntilesN;
    const int m  = lane & 15;
    const int hi = lane >> 4;
    const int krow0 = (kt << 5) + (hi << 3);
    const int n  = (nt << 4) + m;
    _Float16* __restrict__ out = Wp + (((size_t)fi << 5) + lane) * 16;
#pragma unroll
    for (int q = 0; q < 16; ++q)
        out[q] = (_Float16)W[(size_t)(krow0 + ((q >= 8) ? 16 : 0) + (q & 7)) * C + n];
}

// ------------------------------------------------- WMMA GEMM: H = X @ W  (N x 128 @ 128 x 128)
// One wave computes a 16-row M tile across all 8 column tiles.
// A: f32 -> f16 on the fly (two contiguous 32B runs / lane).  B: pre-packed f16 fragments.
__global__ void k_gemm_xw(const float* __restrict__ X, const _Float16* __restrict__ Wp,
                          float* __restrict__ H, int N) {
    const int lane = threadIdx.x & 31;
    const int tile = blockIdx.x * (blockDim.x >> 5) + (threadIdx.x >> 5);
    const int row0 = tile << 4;
    if (row0 >= N) return;
    const int m  = lane & 15;
    const int hi = lane >> 4;          // half-wave: 0 = lanes 0-15, 1 = lanes 16-31
    const int kb = hi << 3;            // A-matrix K sub-offset (0 or 8)
    int arow = row0 + m; if (arow >= N) arow = N - 1;
    const float* __restrict__ xr = X + (size_t)arow * DIMF;
    const v16h* __restrict__ Wv = (const v16h*)Wp;

    v8f acc[8] = {};

    for (int kt = 0; kt < 4; ++kt) {
        const int k0 = kt << 5;
        v16h a;
#pragma unroll
        for (int q = 0; q < 16; ++q) {
            const int k = k0 + kb + ((q >= 8) ? 16 : 0) + (q & 7);
            a[q] = (_Float16)xr[k];
        }
#pragma unroll
        for (int nt = 0; nt < 8; ++nt) {
            const v16h b = Wv[(((kt << 3) + nt) << 5) + lane];
            acc[nt] = __builtin_amdgcn_wmma_f32_16x16x32_f16(
                          false, a, false, b, (short)0, acc[nt], false, false);
        }
    }
    // C/D layout: VGPR r -> row M = r + 8*hi, col N = m
#pragma unroll
    for (int nt = 0; nt < 8; ++nt) {
#pragma unroll
        for (int r = 0; r < 8; ++r) {
            const int rr = row0 + r + (hi << 3);
            if (rr < N) H[(size_t)rr * DIMF + (nt << 4) + m] = acc[nt][r];
        }
    }
}

// ------------------------------------------------- edge scatter: AGG[dst] += H[src]*norm
// One wave per edge, grid-stride. Lane-contiguous float4 gather (b128), prefetch next row.
__global__ void k_scatter(const float* __restrict__ H, const int* __restrict__ src,
                          const int* __restrict__ dst, const float* __restrict__ dinv,
                          float* __restrict__ AGG, int E) {
    const int lane = threadIdx.x & 31;
    const int nw = (gridDim.x * blockDim.x) >> 5;
    int e = (blockIdx.x * blockDim.x + threadIdx.x) >> 5;
    for (; e < E; e += nw) {
        const int s = src[e];
        const int d = dst[e];
        if (e + nw < E) {            // look-ahead: pull next gather row toward L2/L0
            const int s2 = src[e + nw];
            __builtin_prefetch(H + (size_t)s2 * DIMF + (lane << 2), 0, 0);
        }
        const float w = dinv[s] * dinv[d];
        const float4 hv = *(const float4*)(H + (size_t)s * DIMF + (lane << 2));
        float* __restrict__ ad = AGG + (size_t)d * DIMF + (lane << 2);
        atomicAdd(ad + 0, hv.x * w);
        atomicAdd(ad + 1, hv.y * w);
        atomicAdd(ad + 2, hv.z * w);
        atomicAdd(ad + 3, hv.w * w);
    }
}

// ------------------------------------------------- self-loop + bias + LayerNorm + ReLU + residual
// One wave per node row (wave32: float4 per lane, shuffle reductions).
__global__ void k_ln_relu_res(const float* __restrict__ AGG, const float* __restrict__ H,
                              const float* __restrict__ dinv, const float* __restrict__ bias,
                              const float* __restrict__ g, const float* __restrict__ bb,
                              float* __restrict__ X, int N) {
    const int lane = threadIdx.x & 31;
    const int i = blockIdx.x * (blockDim.x >> 5) + (threadIdx.x >> 5);
    if (i >= N) return;
    const float di = dinv[i];
    const float selfw = di * di;
    const int f0 = lane << 2;
    const float4 av = *(const float4*)(AGG + (size_t)i * DIMF + f0);
    const float4 hv = *(const float4*)(H   + (size_t)i * DIMF + f0);
    const float4 bi = *(const float4*)(bias + f0);
    float v[4];
    v[0] = av.x + hv.x * selfw + bi.x;
    v[1] = av.y + hv.y * selfw + bi.y;
    v[2] = av.z + hv.z * selfw + bi.z;
    v[3] = av.w + hv.w * selfw + bi.w;
    float sum = v[0] + v[1] + v[2] + v[3];
#pragma unroll
    for (int off = 16; off > 0; off >>= 1) sum += __shfl_xor(sum, off, 32);
    const float mu = sum * (1.0f / DIMF);
    float var = 0.f;
#pragma unroll
    for (int j = 0; j < 4; ++j) { const float d = v[j] - mu; var += d * d; }
#pragma unroll
    for (int off = 16; off > 0; off >>= 1) var += __shfl_xor(var, off, 32);
    var *= (1.0f / DIMF);
    const float rs = rsqrtf(var + LNEPS);
    const float4 gv = *(const float4*)(g  + f0);
    const float4 bv = *(const float4*)(bb + f0);
    float4 xv = *(float4*)(X + (size_t)i * DIMF + f0);
    xv.x += fmaxf((v[0] - mu) * rs * gv.x + bv.x, 0.0f);
    xv.y += fmaxf((v[1] - mu) * rs * gv.y + bv.y, 0.0f);
    xv.z += fmaxf((v[2] - mu) * rs * gv.z + bv.z, 0.0f);
    xv.w += fmaxf((v[3] - mu) * rs * gv.w + bv.w, 0.0f);
    *(float4*)(X + (size_t)i * DIMF + f0) = xv;
}

// ------------------------------------------------- classifier: OUT = X @ CW + cb  (128 -> 16)
__global__ void k_gemm_cls(const float* __restrict__ X, const _Float16* __restrict__ Wp,
                           const float* __restrict__ cb, float* __restrict__ OUT,
                           int N, int C) {
    const int lane = threadIdx.x & 31;
    const int tile = blockIdx.x * (blockDim.x >> 5) + (threadIdx.x >> 5);
    const int row0 = tile << 4;
    if (row0 >= N) return;
    const int m  = lane & 15;
    const int hi = lane >> 4;
    const int kb = hi << 3;
    int arow = row0 + m; if (arow >= N) arow = N - 1;
    const float* __restrict__ xr = X + (size_t)arow * DIMF;
    const v16h* __restrict__ Wv = (const v16h*)Wp;

    v8f acc = {};
    for (int kt = 0; kt < 4; ++kt) {
        const int k0 = kt << 5;
        v16h a;
#pragma unroll
        for (int q = 0; q < 16; ++q) {
            const int k = k0 + kb + ((q >= 8) ? 16 : 0) + (q & 7);
            a[q] = (_Float16)xr[k];
        }
        const v16h b = Wv[(kt << 5) + lane];
        acc = __builtin_amdgcn_wmma_f32_16x16x32_f16(
                  false, a, false, b, (short)0, acc, false, false);
    }
    const float bias = cb[m];
#pragma unroll
    for (int r = 0; r < 8; ++r) {
        const int rr = row0 + r + (hi << 3);
        if (rr < N) OUT[(size_t)rr * C + m] = acc[r] + bias;
    }
}

// ---------------------------------------------------------------- host launcher
extern "C" void kernel_launch(void* const* d_in, const int* in_sizes, int n_in,
                              void* d_out, int out_size, void* d_ws, size_t ws_size,
                              hipStream_t stream) {
    const float* x      = (const float*)d_in[0];
    const int*   edge   = (const int*)  d_in[1];
    const float* Ws     = (const float*)d_in[2];
    const float* bs     = (const float*)d_in[3];
    const float* ln_g   = (const float*)d_in[4];
    const float* ln_b   = (const float*)d_in[5];
    const float* cls_W  = (const float*)d_in[6];
    const float* cls_b  = (const float*)d_in[7];

    const int N = in_sizes[0] / DIMF;
    const int E = in_sizes[1] / 2;
    const int L = in_sizes[2] / (DIMF * DIMF);
    const int C = in_sizes[7];

    const int* src = edge;
    const int* dst = edge + E;

    // workspace layout (bytes), keep 64B alignment for packed f16 fragments
    char* wsb = (char*)d_ws;
    const size_t ndBytes = (size_t)N * DIMF * sizeof(float);
    float* x_cur = (float*)(wsb);                          // N*128 f32
    float* h     = (float*)(wsb + ndBytes);                // N*128 f32
    float* agg   = (float*)(wsb + 2 * ndBytes);            // N*128 f32
    size_t off   = 3 * ndBytes;
    float* dinv  = (float*)(wsb + off);                    // N f32
    off += (size_t)N * sizeof(float);
    off = (off + 63) & ~(size_t)63;
    _Float16* Wp   = (_Float16*)(wsb + off);               // 32 frag * 32 lanes * 16 f16 = 32KB
    off += (size_t)32 * 32 * 16 * sizeof(_Float16);
    _Float16* Wpc  = (_Float16*)(wsb + off);               // 4 frag * 32 * 16 f16 = 4KB

    const int ND = N * DIMF;
    const int tiles = (N + 15) / 16;
    const int WPB = 8;                              // waves per 256-thread block
    const dim3 blk(256);
    const dim3 grid_tiles((tiles + WPB - 1) / WPB);
    const dim3 grid_nodesw((N + WPB - 1) / WPB);
    const dim3 grid_nd((ND + 255) / 256);
    const dim3 grid_n((N + 255) / 256);
    const dim3 grid_e((E + 255) / 256);
    const dim3 grid_scatter(4096);                  // grid-stride over edges

    // x_cur = x
    hipMemcpyAsync(x_cur, x, ndBytes, hipMemcpyDeviceToDevice, stream);

    // degrees -> dinv
    k_fill<<<grid_n, blk, 0, stream>>>(dinv, 1.0f, N);           // self loop
    k_degree<<<grid_e, blk, 0, stream>>>(dst, dinv, E);
    k_rsqrt_inplace<<<grid_n, blk, 0, stream>>>(dinv, N);

    // pre-pack classifier weights once
    k_pack_b<<<dim3(1), blk, 0, stream>>>(cls_W, Wpc, C, 1);

    for (int l = 0; l < L; ++l) {
        const float* Wl = Ws + (size_t)l * DIMF * DIMF;
        k_pack_b<<<dim3(4), blk, 0, stream>>>(Wl, Wp, DIMF, 8);
        k_gemm_xw<<<grid_tiles, blk, 0, stream>>>(x_cur, Wp, h, N);
        k_fill<<<grid_nd, blk, 0, stream>>>(agg, 0.0f, ND);
        k_scatter<<<grid_scatter, blk, 0, stream>>>(h, src, dst, dinv, agg, E);
        k_ln_relu_res<<<grid_nodesw, blk, 0, stream>>>(
            agg, h, dinv, bs + (size_t)l * DIMF,
            ln_g + (size_t)l * DIMF, ln_b + (size_t)l * DIMF, x_cur, N);
    }

    k_gemm_cls<<<grid_tiles, blk, 0, stream>>>(x_cur, Wpc, cls_b, (float*)d_out, N, C);
}
